// RENBase_14010183319706
// MI455X (gfx1250) — compile-verified, matrix operand order checked by reading, not run on hardware
//
#include <hip/hip_runtime.h>
#include <hip/hip_bf16.h>

typedef __attribute__((ext_vector_type(16))) _Float16 v16h;
typedef __attribute__((ext_vector_type(8)))  float    v8f;
typedef __attribute__((ext_vector_type(4)))  unsigned int u32x4;
typedef __attribute__((ext_vector_type(8)))  int i32x8;
typedef __attribute__((ext_vector_type(4)))  int i32x4;

// Problem dims (from reference)
constexpr int CNU = 96, CNX = 128, CNV = 256, CNY = 96, CT = 128, CB = 512;

// Packed-weight fragment offsets (units of v16h = 32 bytes).
constexpr int OFF_C1  = 0;
constexpr int OFF_D12 = OFF_C1  + (CNV/16)*(CNX/32)*32;   // 2048
constexpr int OFF_D11 = OFF_D12 + (CNV/16)*(CNU/32)*32;   // +1536
constexpr int OFF_A   = OFF_D11 + (CNV/16)*(CNV/32)*32;   // +4096
constexpr int OFF_B1  = OFF_A   + (CNX/16)*(CNX/32)*32;   // +1024
constexpr int OFF_B2  = OFF_B1  + (CNX/16)*(CNV/32)*32;   // +2048
constexpr int OFF_C2  = OFF_B2  + (CNX/16)*(CNU/32)*32;   // +768
constexpr int OFF_D21 = OFF_C2  + (CNY/16)*(CNX/32)*32;   // +768
constexpr int OFF_D22 = OFF_D21 + (CNY/16)*(CNV/32)*32;   // +1536
constexpr int OFF_TOTAL = OFF_D22 + (CNY/16)*(CNU/32)*32; // 14400 v16h = 460800 B

constexpr int NVT_C1 = 2048, NVT_D12 = 1536, NVT_B1 = 2048, NVT_D21 = 1536;

__device__ __forceinline__ v8f splat8(float x) {
  v8f r;
#pragma unroll
  for (int i = 0; i < 8; ++i) r[i] = x;
  return r;
}

__device__ __forceinline__ v8f wmma_acc(v16h a, v16h b, v8f c) {
  return __builtin_amdgcn_wmma_f32_16x16x32_f16(false, a, false, b, (short)0, c,
                                                false, false);
}

// A-fragment (16 rows x 32 K, f16) from an f32 row-major LDS tile.
__device__ __forceinline__ v16h load_a_frag(const float* tile, int stride,
                                            int kbase, int lane) {
  const int m = lane & 15, h = lane >> 4;
  const float* row = tile + m * stride + kbase;
  v16h a;
#pragma unroll
  for (int i = 0; i < 16; i += 2) {
    const int k = (i >> 3) * 16 + h * 8 + (i & 7);
    const float2 f = *(const float2*)(row + k);
    a[i]     = (_Float16)f.x;
    a[i + 1] = (_Float16)f.y;
  }
  return a;
}

// Issue a 1-D Tensor-Data-Mover DMA: nbytes (multiple of 8) global -> LDS.
// D# per CDNA5 ISA ch.8: group0 {count=1, lds_addr, global_addr, type=2};
// group1 {data_size=8B, tensor_dim0 = tile_dim0 = n elements}.
__device__ __forceinline__ void tdm_load_1d(unsigned lds_addr, const void* gptr,
                                            unsigned nbytes) {
  const unsigned long long ga = (unsigned long long)gptr;
  const unsigned n = nbytes >> 3;  // 8-byte elements
  u32x4 g0;
  g0[0] = 1u;                                           // count=1, user mode
  g0[1] = lds_addr;                                     // D#.lds_addr
  g0[2] = (unsigned)(ga & 0xffffffffu);                 // global_addr[31:0]
  g0[3] = (unsigned)((ga >> 32) & 0x01ffffffu) | (2u << 30);  // [56:32] | type=2
  i32x8 g1;
  g1[0] = (int)(3u << 16);                 // data_size=3 (8 bytes), mask=0
  g1[1] = (int)((n & 0xffffu) << 16);      // tensor_dim0[15:0]
  g1[2] = (int)((n >> 16) | (1u << 16));   // tensor_dim0[31:16], tensor_dim1=1
  g1[3] = (int)((n & 0xffffu) << 16);      // tile_dim0 = n
  g1[4] = 0;                               // tile_dim1/2 unused
  g1[5] = (int)n;                          // tensor_dim0_stride = n
  g1[6] = 0;
  g1[7] = 0;
  i32x4 z4 = {0, 0, 0, 0};
  i32x8 z8 = {0, 0, 0, 0, 0, 0, 0, 0};
  __builtin_amdgcn_tensor_load_to_lds(g0, g1, z4, z4, z8, 0);
}

__device__ __forceinline__ unsigned lds_off(const void* p) {
  // Low 32 bits of a generic LDS address are the wave-relative LDS offset.
  return (unsigned)(unsigned long long)p;
}

// ---- prep: pack one weight matrix into B-fragment layout (f16) --------------
__global__ void pack_w_kernel(const float* __restrict__ W, v16h* __restrict__ dst,
                              int N, int K, int d11mode) {
  const int ktiles = K / 32;
  const int total  = (N / 16) * ktiles * 32;
  const int idx = blockIdx.x * blockDim.x + threadIdx.x;
  if (idx >= total) return;
  const int lane = idx & 31;
  const int t    = idx >> 5;
  const int kt   = t % ktiles;
  const int nt   = t / ktiles;
  const int n = nt * 16 + (lane & 15);
  const int h = lane >> 4;
  v16h v;
#pragma unroll
  for (int i = 0; i < 16; ++i) {
    const int k = kt * 32 + (i >> 3) * 16 + h * 8 + (i & 7);
    float val = W[n * K + k];
    if (d11mode && ((k >> 4) >= (n >> 4))) val = 0.f;  // strict block-lower only
    v[i] = (_Float16)val;
  }
  dst[idx] = v;
}

// ---- prep: D11 diagonal 16x16 blocks, strict-lower, f32 --------------------
__global__ void extract_diag_kernel(const float* __restrict__ D11,
                                    float* __restrict__ dst) {
  const int idx = blockIdx.x * blockDim.x + threadIdx.x;
  if (idx >= 16 * 16 * 16) return;
  const int b = idx >> 8, n = (idx >> 4) & 15, k = idx & 15;
  dst[idx] = (n > k) ? D11[(b * 16 + n) * CNV + (b * 16 + k)] : 0.f;
}

// ---- main persistent rollout kernel: 1 wave32 per 16-row batch tile --------
__global__ __launch_bounds__(32, 1) void ren_rollout_kernel(
    const float* __restrict__ x0, const float* __restrict__ u,
    const float* __restrict__ bxg, const float* __restrict__ bvg,
    const float* __restrict__ byg, const v16h* __restrict__ pk,
    const float* __restrict__ diag, float* __restrict__ out) {
  __shared__ float s_x[2][16 * CNX];   // double-buffered state tile (16 KB)
  __shared__ float s_w[16 * CNV];      // equilibrium tile (16 KB)
  __shared__ float s_u[2][16 * CNU];   // double-buffered input tile (12 KB)
  __shared__ float s_diag[16 * 16 * 16];
  __shared__ float s_bx[CNX], s_bv[CNV], s_by[CNY];
  // LDS-cached packed weight fragments (hottest 224 KB).
  __shared__ v16h s_c1[NVT_C1];
  __shared__ v16h s_d12[NVT_D12];
  __shared__ v16h s_b1[NVT_B1];
  __shared__ v16h s_d21[NVT_D21];

  const int lane = threadIdx.x;
  const int row0 = blockIdx.x * 16;
  const int nlo  = lane & 15;
  const int hi16 = lane & 16;
  const int mrow = (lane >> 4) * 8;

  // Kick off TDM DMAs: weight fragment caches + first u tile.  These stream
  // while the wave does its scalar init below; retired by s_wait_tensorcnt.
  if (lane == 0) {
    tdm_load_1d(lds_off(s_c1),  pk + OFF_C1,  NVT_C1  * (int)sizeof(v16h));
    tdm_load_1d(lds_off(s_d12), pk + OFF_D12, NVT_D12 * (int)sizeof(v16h));
    tdm_load_1d(lds_off(s_b1),  pk + OFF_B1,  NVT_B1  * (int)sizeof(v16h));
    tdm_load_1d(lds_off(s_d21), pk + OFF_D21, NVT_D21 * (int)sizeof(v16h));
    tdm_load_1d(lds_off(&s_u[0][0]), u + (size_t)row0 * CNU, 16 * CNU * 4);
  }

  for (int i = lane; i < 4096; i += 32) s_diag[i] = diag[i];
  for (int i = lane; i < CNX; i += 32) s_bx[i] = bxg[i];
  for (int i = lane; i < CNV; i += 32) s_bv[i] = bvg[i];
  for (int i = lane; i < CNY; i += 32) s_by[i] = byg[i];
  for (int i = lane; i < 16 * CNX; i += 32) s_x[0][i] = x0[row0 * CNX + i];
  for (int i = lane; i < 16 * CNV; i += 32) s_w[i] = 0.f;  // tril padding

#pragma unroll 1
  for (int t = 0; t < CT; ++t) {
    const int cur = t & 1, nxt = cur ^ 1;

    __builtin_amdgcn_s_wait_tensorcnt(0);  // u[t] (and, at t=0, weights) ready

    // Hoisted A-fragments of x (K=128) and u (K=96).
    v16h xf[4], uf[3];
#pragma unroll
    for (int c = 0; c < 4; ++c) xf[c] = load_a_frag(&s_x[cur][0], CNX, 32 * c, lane);
#pragma unroll
    for (int c = 0; c < 3; ++c) uf[c] = load_a_frag(&s_u[cur][0], CNU, 32 * c, lane);

    // Overlap next step's input DMA with this step's compute.
    if (t + 1 < CT && lane == 0)
      tdm_load_1d(lds_off(&s_u[nxt][0]),
                  u + ((size_t)(t + 1) * CB + row0) * CNU, 16 * CNU * 4);

    // Blocked forward substitution over NV = 16 blocks of 16.
#pragma unroll 1
    for (int bi = 0; bi < 16; ++bi) {
      v8f acc = splat8(s_bv[bi * 16 + nlo]);
#pragma unroll
      for (int c = 0; c < 4; ++c)
        acc = wmma_acc(xf[c], s_c1[(bi * 4 + c) * 32 + lane], acc);
#pragma unroll
      for (int c = 0; c < 3; ++c)
        acc = wmma_acc(uf[c], s_d12[(bi * 3 + c) * 32 + lane], acc);

      // Off-diagonal D11 (packed weights zero on/above the block diagonal, so
      // the odd-bi half-chunk overlap contributes exactly zero).
      const int nc = (bi + 1) >> 1;
      for (int c = 0; c < nc; ++c) {
        v16h wfrag = load_a_frag(s_w, CNV, 32 * c, lane);
        acc = wmma_acc(wfrag, pk[OFF_D11 + (bi * 8 + c) * 32 + lane], acc);
      }

      // In-register 16-round solve of the strictly-lower diagonal block.
      float dcol[16];
      const float* Dc = s_diag + bi * 256 + nlo * 16;
#pragma unroll
      for (int k = 0; k < 16; ++k) dcol[k] = Dc[k];
#pragma unroll
      for (int k = 0; k < 16; ++k) {
#pragma unroll
        for (int r = 0; r < 8; ++r) {
          float wk = __shfl(acc[r], k + hi16, 32);
          wk = fmaxf(wk, 0.f);
          acc[r] = fmaf(wk, dcol[k], acc[r]);  // dcol[k]==0 for rows <= k
        }
      }
#pragma unroll
      for (int r = 0; r < 8; ++r)
        s_w[(r + mrow) * CNV + bi * 16 + nlo] = fmaxf(acc[r], 0.f);
    }

    v16h wf[8];
#pragma unroll
    for (int c = 0; c < 8; ++c) wf[c] = load_a_frag(s_w, CNV, 32 * c, lane);

    // y_t -> global
    float* yg = out + (size_t)CB * CNX + ((size_t)t * CB + row0) * CNY;
#pragma unroll 1
    for (int nt = 0; nt < 6; ++nt) {
      v8f acc = splat8(s_by[nt * 16 + nlo]);
#pragma unroll
      for (int c = 0; c < 4; ++c)
        acc = wmma_acc(xf[c], pk[OFF_C2 + (nt * 4 + c) * 32 + lane], acc);
#pragma unroll
      for (int c = 0; c < 8; ++c)
        acc = wmma_acc(wf[c], s_d21[(nt * 8 + c) * 32 + lane], acc);
#pragma unroll
      for (int c = 0; c < 3; ++c)
        acc = wmma_acc(uf[c], pk[OFF_D22 + (nt * 3 + c) * 32 + lane], acc);
#pragma unroll
      for (int r = 0; r < 8; ++r)
        yg[(r + mrow) * CNY + nt * 16 + nlo] = acc[r];
    }

    // x_{t+1} -> other LDS buffer
#pragma unroll 1
    for (int nt = 0; nt < 8; ++nt) {
      v8f acc = splat8(s_bx[nt * 16 + nlo]);
#pragma unroll
      for (int c = 0; c < 4; ++c)
        acc = wmma_acc(xf[c], pk[OFF_A + (nt * 4 + c) * 32 + lane], acc);
#pragma unroll
      for (int c = 0; c < 8; ++c)
        acc = wmma_acc(wf[c], s_b1[(nt * 8 + c) * 32 + lane], acc);
#pragma unroll
      for (int c = 0; c < 3; ++c)
        acc = wmma_acc(uf[c], pk[OFF_B2 + (nt * 3 + c) * 32 + lane], acc);
#pragma unroll
      for (int r = 0; r < 8; ++r)
        s_x[nxt][(r + mrow) * CNX + nt * 16 + nlo] = acc[r];
    }
  }

  // T=128 even -> final state lives in buffer 0.
  for (int i = lane; i < 16 * CNX; i += 32) out[row0 * CNX + i] = s_x[0][i];
}

extern "C" void kernel_launch(void* const* d_in, const int* in_sizes, int n_in,
                              void* d_out, int out_size, void* d_ws, size_t ws_size,
                              hipStream_t stream) {
  (void)in_sizes; (void)n_in; (void)out_size; (void)ws_size;
  const float* x0  = (const float*)d_in[0];
  const float* u   = (const float*)d_in[1];
  const float* A   = (const float*)d_in[2];
  const float* B1  = (const float*)d_in[3];
  const float* B2  = (const float*)d_in[4];
  const float* C1  = (const float*)d_in[5];
  const float* C2  = (const float*)d_in[6];
  const float* D11 = (const float*)d_in[7];
  const float* D12 = (const float*)d_in[8];
  const float* D21 = (const float*)d_in[9];
  const float* D22 = (const float*)d_in[10];
  const float* bx  = (const float*)d_in[11];
  const float* bv  = (const float*)d_in[12];
  const float* by  = (const float*)d_in[13];

  v16h*  pk   = (v16h*)d_ws;
  float* diag = (float*)((char*)d_ws + sizeof(v16h) * (size_t)OFF_TOTAL);

  auto pack = [&](const float* W, int off, int N, int K, int d11mode) {
    const int total = (N / 16) * (K / 32) * 32;
    pack_w_kernel<<<(total + 255) / 256, 256, 0, stream>>>(W, pk + off, N, K, d11mode);
  };
  pack(C1,  OFF_C1,  CNV, CNX, 0);
  pack(D12, OFF_D12, CNV, CNU, 0);
  pack(D11, OFF_D11, CNV, CNV, 1);
  pack(A,   OFF_A,   CNX, CNX, 0);
  pack(B1,  OFF_B1,  CNX, CNV, 0);
  pack(B2,  OFF_B2,  CNX, CNU, 0);
  pack(C2,  OFF_C2,  CNY, CNX, 0);
  pack(D21, OFF_D21, CNY, CNV, 0);
  pack(D22, OFF_D22, CNY, CNU, 0);
  extract_diag_kernel<<<(4096 + 255) / 256, 256, 0, stream>>>(D11, diag);

  ren_rollout_kernel<<<CB / 16, 32, 0, stream>>>(x0, u, bx, bv, by, pk, diag,
                                                 (float*)d_out);
}
